// LSTM_block_24644522344592
// MI455X (gfx1250) — matrix operand order repaired and checked by reading, not compiled, not run on hardware
//
#include <hip/hip_runtime.h>
#include <hip/hip_bf16.h>

// ---------------------------------------------------------------------------
// LSTM on gfx1250 (MI455X): bf16 WMMA for both the input projection GEMM and
// the sequential recurrence. Batch rows are independent in the recurrence, so
// 4 persistent workgroups each own 16 batch rows; h lives in LDS (bf16,
// double-buffered), c lives in accumulator VGPRs. The recurrence K-loop is
// software-pipelined (double-buffered A/B fragments) to hide L2 latency.
// ---------------------------------------------------------------------------

typedef __attribute__((ext_vector_type(16))) __bf16    v16bf;
typedef __attribute__((ext_vector_type(8)))  float     v8f;
typedef __attribute__((ext_vector_type(4)))  unsigned  u32x4;

union Frag { v16bf v; u32x4 q[2]; };

#define B_DIM   64
#define S_DIM   512
#define I_DIM   256
#define H_DIM   512
#define G_DIM   2048   // 4*H

__device__ __forceinline__ unsigned short f2bf_bits(float f) {
    union { float f; unsigned u; } v; v.f = f;
    unsigned r = v.u + 0x7FFFu + ((v.u >> 16) & 1u);   // round-to-nearest-even
    return (unsigned short)(r >> 16);
}

__device__ __forceinline__ float sigmoid_(float x) {
    return 1.0f / (1.0f + __expf(-x));
}
__device__ __forceinline__ float tanh_(float x) {
    return 2.0f * sigmoid_(2.0f * x) - 1.0f;
}

// ---------------------------------------------------------------------------
// Phase 0: convert x / W_ih / W_hh to bf16, fuse the two biases.
// ---------------------------------------------------------------------------
#define XN   (B_DIM * S_DIM * I_DIM)     // 8388608
#define WIHN (G_DIM * I_DIM)             // 524288
#define WHHN (G_DIM * H_DIM)             // 1048576
#define TOTAL_PREP (XN + WIHN + WHHN + G_DIM)

__global__ void __launch_bounds__(256)
lstm_prep(const float* __restrict__ x, const float* __restrict__ wih,
          const float* __restrict__ whh, const float* __restrict__ bih,
          const float* __restrict__ bhh,
          unsigned short* __restrict__ xbf, unsigned short* __restrict__ wihbf,
          unsigned short* __restrict__ whhbf, float* __restrict__ bias) {
    for (long i = (long)blockIdx.x * blockDim.x + threadIdx.x; i < TOTAL_PREP;
         i += (long)gridDim.x * blockDim.x) {
        if (i < XN) {
            xbf[i] = f2bf_bits(x[i]);
        } else if (i < XN + WIHN) {
            long j = i - XN;
            wihbf[j] = f2bf_bits(wih[j]);
        } else if (i < XN + WIHN + WHHN) {
            long j = i - XN - WIHN;
            whhbf[j] = f2bf_bits(whh[j]);
        } else {
            long j = i - XN - WIHN - WHHN;
            bias[j] = bih[j] + bhh[j];
        }
    }
}

// ---------------------------------------------------------------------------
// Phase 1: x_proj[32768 x 2048] = Xbf[32768 x 256] @ Wih^T + bias   (bf16 WMMA)
// Block = 128 threads (4 waves); each wave: one 16-row M tile x 64 N cols.
// ---------------------------------------------------------------------------
__global__ void __launch_bounds__(128)
lstm_xproj(const unsigned short* __restrict__ xbf,
           const unsigned short* __restrict__ wihbf,
           const float* __restrict__ bias, float* __restrict__ xproj) {
    const int wave  = threadIdx.x >> 5;
    const int lane  = threadIdx.x & 31;
    const int khalf = lane >> 4;
    const int l16   = lane & 15;

    const int Mbase = blockIdx.y * 64 + wave * 16;
    const int Nbase = blockIdx.x * 64;

    const unsigned short* arow = xbf + (size_t)(Mbase + l16) * I_DIM;

    v8f acc[4] = {};

    for (int k0 = 0; k0 < I_DIM; k0 += 32) {
        Frag a;
        a.q[0] = *(const u32x4*)(arow + k0 + khalf * 8);
        a.q[1] = *(const u32x4*)(arow + k0 + 16 + khalf * 8);
#pragma unroll
        for (int nt = 0; nt < 4; ++nt) {
            const int ncol = Nbase + nt * 16 + l16;
            const unsigned short* brow =
                wihbf + (size_t)ncol * I_DIM + k0 + khalf * 8;
            Frag b;
            b.q[0] = *(const u32x4*)(brow);
            b.q[1] = *(const u32x4*)(brow + 16);
            acc[nt] = __builtin_amdgcn_wmma_f32_16x16x32_bf16(
                false, a.v, false, b.v, (short)0, acc[nt], false, false);
        }
    }

#pragma unroll
    for (int nt = 0; nt < 4; ++nt) {
        const int col = Nbase + nt * 16 + l16;
        const float bv = bias[col];
#pragma unroll
        for (int r = 0; r < 8; ++r) {
            const int row = Mbase + khalf * 8 + r;
            xproj[(size_t)row * G_DIM + col] = acc[nt][r] + bv;
        }
    }
}

// ---------------------------------------------------------------------------
// Phase 2: recurrence. 4 blocks x 1024 threads (32 waves). Block bg owns
// batch rows [bg*16, bg*16+16). Wave w owns h columns [w*16, w*16+16) and
// computes the 4 gate tiles (i,f,g,o) for those columns each step.
// Software-pipelined K loop: stage it+1's fragments while WMMA-ing stage it.
// ---------------------------------------------------------------------------
__global__ void __launch_bounds__(1024)
lstm_recur(const unsigned short* __restrict__ whhbf,
           const float* __restrict__ xproj, float* __restrict__ out) {
    __shared__ unsigned short hbuf[2][16 * H_DIM];   // 32 KB

    const int tid   = threadIdx.x;
    const int wave  = tid >> 5;
    const int lane  = tid & 31;
    const int khalf = lane >> 4;
    const int l16   = lane & 15;
    const int bg    = blockIdx.x;                    // batch group 0..3

    const long HOFF = (long)B_DIM * S_DIM * H_DIM;   // start of h_final
    const long COFF = HOFF + (long)B_DIM * H_DIM;    // start of c_final

    // h0 = 0
    for (int i = tid; i < 16 * H_DIM; i += 1024) hbuf[0][i] = 0;
    __syncthreads();

    v8f cst = {};                                    // c state (16x16 tile)
    const int hcol = wave * 16 + l16;                // this lane's h column

    // Per-gate W_hh fragment base pointers (step-invariant, hoisted).
    const unsigned short* bbase[4];
#pragma unroll
    for (int g = 0; g < 4; ++g)
        bbase[g] = whhbf + (size_t)(g * H_DIM + hcol) * H_DIM + khalf * 8;

    // Per-r x_proj row bases (advance by G_DIM each step).
    size_t xrow[8];
#pragma unroll
    for (int r = 0; r < 8; ++r)
        xrow[r] = (size_t)(bg * 16 + khalf * 8 + r) * S_DIM * G_DIM + hcol;

    int p = 0;

    for (int s = 0; s < S_DIM; ++s) {
        // Issue this step's x_proj loads up front (single-use -> nontemporal),
        // hidden behind the 64 WMMAs of the K loop below.
        float xp[4][8];
#pragma unroll
        for (int r = 0; r < 8; ++r) {
#pragma unroll
            for (int g = 0; g < 4; ++g)
                xp[g][r] = __builtin_nontemporal_load(
                    &xproj[xrow[r] + (size_t)s * G_DIM + g * H_DIM]);
        }
        // Prefetch next step's x_proj (global_prefetch_b8).
        if (s + 1 < S_DIM)
            __builtin_prefetch(&xproj[xrow[0] + (size_t)(s + 1) * G_DIM], 0, 1);

        const unsigned short* hrow = &hbuf[p][l16 * H_DIM];

        // ---- software-pipelined GEMM: gates += h @ W_hh^T ----
        Frag a[2], b[2][4];
        a[0].q[0] = *(const u32x4*)(hrow + khalf * 8);
        a[0].q[1] = *(const u32x4*)(hrow + 16 + khalf * 8);
#pragma unroll
        for (int g = 0; g < 4; ++g) {
            b[0][g].q[0] = *(const u32x4*)(bbase[g]);
            b[0][g].q[1] = *(const u32x4*)(bbase[g] + 16);
        }

        v8f acc[4] = {};
#pragma unroll
        for (int it = 0; it < 16; ++it) {
            const int cur = it & 1, nxt = cur ^ 1;
            if (it < 15) {
                const int kk = (it + 1) * 32;
                a[nxt].q[0] = *(const u32x4*)(hrow + kk + khalf * 8);
                a[nxt].q[1] = *(const u32x4*)(hrow + kk + 16 + khalf * 8);
#pragma unroll
                for (int g = 0; g < 4; ++g) {
                    b[nxt][g].q[0] = *(const u32x4*)(bbase[g] + kk);
                    b[nxt][g].q[1] = *(const u32x4*)(bbase[g] + kk + 16);
                }
            }
#pragma unroll
            for (int g = 0; g < 4; ++g)
                acc[g] = __builtin_amdgcn_wmma_f32_16x16x32_bf16(
                    false, a[cur].v, false, b[cur][g].v, (short)0, acc[g],
                    false, false);
        }

        // ---- gate math + state update for this wave's 16x16 tile ----
#pragma unroll
        for (int r = 0; r < 8; ++r) {
            const int lrow = khalf * 8 + r;          // local batch row
            const int grow = bg * 16 + lrow;         // global batch row

            const float iv = sigmoid_(acc[0][r] + xp[0][r]);
            const float fv = sigmoid_(acc[1][r] + xp[1][r]);
            const float gv = tanh_   (acc[2][r] + xp[2][r]);
            const float ov = sigmoid_(acc[3][r] + xp[3][r]);

            const float cn = fv * cst[r] + iv * gv;
            cst[r] = cn;
            const float hn = ov * tanh_(cn);

            __builtin_nontemporal_store(
                hn, &out[((long)grow * S_DIM + s) * H_DIM + hcol]);
            hbuf[p ^ 1][lrow * H_DIM + hcol] = f2bf_bits(hn);

            if (s == S_DIM - 1) {
                out[HOFF + (long)grow * H_DIM + hcol] = hn;
                out[COFF + (long)grow * H_DIM + hcol] = cn;
            }
        }
        __syncthreads();
        p ^= 1;
    }
}

// ---------------------------------------------------------------------------
// Host launcher
// ---------------------------------------------------------------------------
extern "C" void kernel_launch(void* const* d_in, const int* in_sizes, int n_in,
                              void* d_out, int out_size, void* d_ws, size_t ws_size,
                              hipStream_t stream) {
    const float* x    = (const float*)d_in[0];
    const float* W_ih = (const float*)d_in[1];
    const float* W_hh = (const float*)d_in[2];
    const float* b_ih = (const float*)d_in[3];
    const float* b_hh = (const float*)d_in[4];
    float* out = (float*)d_out;

    // Workspace layout (all 16B aligned):
    unsigned short* xbf   = (unsigned short*)d_ws;
    unsigned short* wihbf = xbf + XN;
    unsigned short* whhbf = wihbf + WIHN;
    float*          bias  = (float*)(whhbf + WHHN);
    float*          xproj = bias + G_DIM;

    lstm_prep<<<2048, 256, 0, stream>>>(x, W_ih, W_hh, b_ih, b_hh,
                                        xbf, wihbf, whhbf, bias);

    // x_proj GEMM: grid = (2048/64 N-tiles, 32768/64 M-tiles)
    lstm_xproj<<<dim3(G_DIM / 64, (B_DIM * S_DIM) / 64), 128, 0, stream>>>(
        xbf, wihbf, bias, xproj);

    // Recurrence: 4 persistent blocks, one per 16-row batch group.
    lstm_recur<<<4, 1024, 0, stream>>>(whhbf, xproj, out);
}
